// FNOGNO_17884243820666
// MI455X (gfx1250) — matrix operand order repaired and checked by reading, not compiled, not run on hardware
//
#include <hip/hip_runtime.h>
#include <math.h>

// ---------------------------------------------------------------------------
// FNO-GNO forward for MI455X (gfx1250, wave32).
// bf16-resident pipeline: all activations + pre-transposed weights live in
// workspace as bf16.  GEMMs use v_wmma_f32_16x16x32_bf16 with double-buffered
// LDS tiles filled by CDNA5 async-to-LDS copies (global_load_async_to_lds_b128
// + partial s_wait_asynccnt) so copies for tile k+1 overlap tile-k math.
// Spectral conv = truncated separable DFTs (16x16x8 corner modes) on VALU.
// ---------------------------------------------------------------------------

#define GRID3   262144      // 64^3
#define NOUT    40000
#define KNB     10
#define TWO_PI  6.2831853071795864f

typedef __attribute__((ext_vector_type(16))) __bf16 v16bf;
typedef __attribute__((ext_vector_type(8)))  float  v8f;

__device__ __forceinline__ __bf16 f2bf(float x) {
  union { float f; unsigned u; } v; v.f = x;
  unsigned r = v.u + 0x7FFFu + ((v.u >> 16) & 1u);   // round-to-nearest-even
  union { unsigned short s; __bf16 b; } o; o.s = (unsigned short)(r >> 16);
  return o.b;
}
__device__ __forceinline__ float bf2f(__bf16 b) {
  union { __bf16 b; unsigned short s; } i; i.b = b;
  union { unsigned u; float f; } o; o.u = ((unsigned)i.s) << 16;
  return o.f;
}
__device__ __forceinline__ float gelu_f(float x) {
  return 0.5f * x * (1.0f + tanhf(0.7978845608f * (x + 0.044715f * x * x * x)));
}

// CDNA5 async copy: 16 bytes global -> LDS, tracked by ASYNCcnt.
// LDS address = low 32 bits of the flat pointer (ISA 10.2).
__device__ __forceinline__ void async_copy16(void* lds_ptr, const void* g_ptr) {
  unsigned loff = (unsigned)(unsigned long long)lds_ptr;
  asm volatile("global_load_async_to_lds_b128 %0, %1, off"
               :: "v"(loff), "v"(g_ptr) : "memory");
}
// wait until ASYNCcnt <= N (async loads complete in order, ISA 08 §4.1)
template <int N>
__device__ __forceinline__ void async_wait_le() {
  asm volatile("s_wait_asynccnt %0" :: "i"(N) : "memory");
}

// ---------------------------------------------------------------------------
// WMMA bf16 GEMM:  C[M,N] = act( A[M,K] @ W[K,N] + bias + (Res?) )
// A: bf16 [M][K] row-major.  Wt: bf16 [N][K] (pre-transposed weight).
// 128 threads = 4 waves; block tile 64 x (16*NT); each wave: one M-tile,
// NT N-tiles (A fragment reused across NT wmmas).  Double-buffered LDS,
// async copies per wave per K-step: 2 (A) + NT/2 (B), all unconditional.
// Requires K%32==0, M%64==0, N%(16*NT)==0.
// Fragment layouts per CDNA5 ISA 7.12.2 (16-bit A 16x32 / B 32x16).
// ---------------------------------------------------------------------------
template <int ACT, int NT>
__global__ __launch_bounds__(128)
void gemm_wmma_bf16(const __bf16* __restrict__ A, const __bf16* __restrict__ Wt,
                    const float* __restrict__ bias, const float* __restrict__ Res,
                    __bf16* __restrict__ C, int M, int N, int K) {
  __shared__ alignas(16) __bf16 Bl[2][NT * 16 * 32];
  __shared__ alignas(16) __bf16 Al[2][4][16 * 32];
  constexpr int NCOPY = 2 + NT / 2;   // async instructions per wave per tile

  const int tid  = threadIdx.x;
  const int wv   = tid >> 5;
  const int lane = tid & 31;
  const int n0   = blockIdx.x * (16 * NT);
  const int m0w  = blockIdx.y * 64 + wv * 16;

  // A tile: 16 rows x 4 quads(16B); lane covers quads {lane, lane+32}
  const int ra0 = lane >> 2,       ka0 = (lane & 3) * 8;
  const int ra1 = (lane >> 2) + 8, ka1 = ka0;
  const __bf16* gA0 = A + (size_t)(m0w + ra0) * K + ka0;
  const __bf16* gA1 = A + (size_t)(m0w + ra1) * K + ka1;
  // B tile: NT*16 rows x 4 quads, spread over 4 waves (NT/2 quads per lane)
  int rb[NT / 2], kb[NT / 2];
  const __bf16* gB[NT / 2];
  #pragma unroll
  for (int i = 0; i < NT / 2; ++i) {
    int q = wv * (NT * 16) + i * 32 + lane;
    rb[i] = q >> 2;
    kb[i] = (q & 3) * 8;
    gB[i] = Wt + (size_t)(n0 + rb[i]) * K + kb[i];
  }

  const v8f vzero = {0.f, 0.f, 0.f, 0.f, 0.f, 0.f, 0.f, 0.f};
  v8f acc[NT];
  #pragma unroll
  for (int j = 0; j < NT; ++j) acc[j] = vzero;

  const int KT = K >> 5;
  auto issue = [&](int buf, int k0) {
    async_copy16(&Al[buf][wv][ra0 * 32 + ka0], gA0 + k0);
    async_copy16(&Al[buf][wv][ra1 * 32 + ka1], gA1 + k0);
    #pragma unroll
    for (int i = 0; i < NT / 2; ++i)
      async_copy16(&Bl[buf][rb[i] * 32 + kb[i]], gB[i] + k0);
  };

  issue(0, 0);
  for (int kt = 0; kt < KT; ++kt) {
    const int buf = kt & 1;
    if (kt + 1 < KT) {
      issue(buf ^ 1, (kt + 1) << 5);   // overlap next-tile copies with math
      async_wait_le<NCOPY>();          // this tile's copies (oldest) are done
    } else {
      async_wait_le<0>();
    }
    __syncthreads();                   // all waves' B contributions visible

    const int half = lane >> 4;        // 0: lanes 0-15, 1: lanes 16-31
    const int mrow = lane & 15;        // A row / B column
    union { v16bf v; uint4 q[2]; } fa;
    const __bf16* ap = &Al[buf][wv][mrow * 32 + half * 8];
    fa.q[0] = *(const uint4*)(ap);
    fa.q[1] = *(const uint4*)(ap + 16);
    #pragma unroll
    for (int j = 0; j < NT; ++j) {
      union { v16bf v; uint4 q[2]; } fb;
      const __bf16* bp = &Bl[buf][(j * 16 + mrow) * 32 + half * 8];
      fb.q[0] = *(const uint4*)(bp);
      fb.q[1] = *(const uint4*)(bp + 16);
      acc[j] = __builtin_amdgcn_wmma_f32_16x16x32_bf16(false, fa.v, false, fb.v,
                                                       (short)0, acc[j], false, false);
    }
    __syncthreads();                   // protect buf before it is re-filled
  }

  #pragma unroll
  for (int j = 0; j < NT; ++j) {
    const int nc = n0 + j * 16 + (lane & 15);
    const float bv = bias[nc];
    #pragma unroll
    for (int r = 0; r < 8; ++r) {
      int m = m0w + r + 8 * (lane >> 4);
      size_t off2 = (size_t)m * N + nc;
      float v = acc[j][r] + bv;
      if (Res) v += Res[off2];
      if (ACT == 1) v = gelu_f(v);
      C[off2] = f2bf(v);
    }
  }
}

// transpose + convert + K-pad weight: dst[n*Kp+k] = (k<Ks) ? src[k*N+n] : 0
__global__ void wt_convert(const float* __restrict__ src, __bf16* __restrict__ dst,
                           int Ks, int Kp, int N) {
  int t = blockIdx.x * blockDim.x + threadIdx.x;
  if (t >= N * Kp) return;
  int k = t % Kp, n = t / Kp;
  dst[t] = (k < Ks) ? f2bf(src[(size_t)k * N + n]) : f2bf(0.f);
}

// lift input, K-padded to 32: [f(3) | in_p(3) | zeros(26)]
__global__ void pack_lift_in(const float* __restrict__ f, const float* __restrict__ inp,
                             __bf16* __restrict__ out, int n) {
  int t = blockIdx.x * blockDim.x + threadIdx.x;
  if (t >= n * 32) return;
  int k = t & 31, i = t >> 5;
  float v = 0.f;
  if (k < 3) v = f[(size_t)i * 3 + k];
  else if (k < 6) v = inp[(size_t)i * 3 + k - 3];
  out[t] = f2bf(v);
}

// ---------------- spectral conv: truncated separable DFTs ------------------
__global__ void spec_fz(const __bf16* __restrict__ X, float* __restrict__ T1) {
  int t = blockIdx.x * blockDim.x + threadIdx.x;
  if (t >= 4096 * 8 * 64) return;
  int c = t & 63, kz = (t >> 6) & 7, p = t >> 9;
  float re = 0.f, im = 0.f;
  for (int z = 0; z < 64; ++z) {
    float v = bf2f(X[((size_t)(p * 64 + z)) * 64 + c]);
    float ang = -TWO_PI * (float)((kz * z) & 63) * (1.f / 64.f);
    re += v * cosf(ang);
    im += v * sinf(ang);
  }
  T1[(size_t)t * 2 + 0] = re;
  T1[(size_t)t * 2 + 1] = im;
}

__global__ void spec_fy(const float* __restrict__ T1, float* __restrict__ T2) {
  int t = blockIdx.x * blockDim.x + threadIdx.x;
  if (t >= 64 * 16 * 8 * 64) return;
  int c = t & 63, kz = (t >> 6) & 7, kyi = (t >> 9) & 15, x = t >> 13;
  int ky = (kyi < 8) ? kyi : kyi + 48;
  float re = 0.f, im = 0.f;
  for (int y = 0; y < 64; ++y) {
    const float* s = &T1[(((size_t)(x * 64 + y) * 8 + kz) * 64 + c) * 2];
    float ang = -TWO_PI * (float)((ky * y) & 63) * (1.f / 64.f);
    float cs = cosf(ang), sn = sinf(ang);
    re += s[0] * cs - s[1] * sn;
    im += s[0] * sn + s[1] * cs;
  }
  T2[(size_t)t * 2 + 0] = re;
  T2[(size_t)t * 2 + 1] = im;
}

__global__ void spec_fx(const float* __restrict__ T2, float* __restrict__ T3) {
  int t = blockIdx.x * blockDim.x + threadIdx.x;
  if (t >= 16 * 16 * 8 * 64) return;
  int c = t & 63, kz = (t >> 6) & 7, kyi = (t >> 9) & 15, kxi = t >> 13;
  int kx = (kxi < 8) ? kxi : kxi + 48;
  float re = 0.f, im = 0.f;
  for (int x = 0; x < 64; ++x) {
    const float* s = &T2[(((size_t)(x * 16 + kyi) * 8 + kz) * 64 + c) * 2];
    float ang = -TWO_PI * (float)((kx * x) & 63) * (1.f / 64.f);
    float cs = cosf(ang), sn = sinf(ang);
    re += s[0] * cs - s[1] * sn;
    im += s[0] * sn + s[1] * cs;
  }
  T3[(size_t)t * 2 + 0] = re;
  T3[(size_t)t * 2 + 1] = im;
}

__global__ void spec_modemul(const float* __restrict__ T3, const float* __restrict__ sw,
                             float* __restrict__ OF) {
  int t = blockIdx.x * blockDim.x + threadIdx.x;
  if (t >= 16 * 16 * 8 * 64) return;
  int d = t & 63, kz = (t >> 6) & 7, kyi = (t >> 9) & 15, kxi = t >> 13;
  int corner = ((kxi >= 8) ? 2 : 0) + ((kyi >= 8) ? 1 : 0);
  int kxp = kxi & 7, kyp = kyi & 7;
  const float* blk = &T3[((size_t)(kxi * 16 + kyi) * 8 + kz) * 64 * 2];
  float ore = 0.f, oim = 0.f;
  for (int c = 0; c < 64; ++c) {
    float br = blk[c * 2], bi = blk[c * 2 + 1];
    size_t ir = (((((size_t)(corner * 2 + 0) * 64 + c) * 64 + d) * 8 + kxp) * 8 + kyp) * 8 + kz;
    size_t ii = (((((size_t)(corner * 2 + 1) * 64 + c) * 64 + d) * 8 + kxp) * 8 + kyp) * 8 + kz;
    ore += br * sw[ir] - bi * sw[ii];
    oim += br * sw[ii] + bi * sw[ir];
  }
  OF[(size_t)t * 2 + 0] = ore;
  OF[(size_t)t * 2 + 1] = oim;
}

__global__ void spec_ix(const float* __restrict__ OF, float* __restrict__ U1) {
  int t = blockIdx.x * blockDim.x + threadIdx.x;
  if (t >= 64 * 16 * 8 * 64) return;
  int d = t & 63, kz = (t >> 6) & 7, kyi = (t >> 9) & 15, x = t >> 13;
  float re = 0.f, im = 0.f;
  for (int kxi = 0; kxi < 16; ++kxi) {
    int kx = (kxi < 8) ? kxi : kxi + 48;
    const float* s = &OF[(((size_t)(kxi * 16 + kyi) * 8 + kz) * 64 + d) * 2];
    float ang = TWO_PI * (float)((kx * x) & 63) * (1.f / 64.f);
    float cs = cosf(ang), sn = sinf(ang);
    re += s[0] * cs - s[1] * sn;
    im += s[0] * sn + s[1] * cs;
  }
  U1[(size_t)t * 2 + 0] = re * (1.f / 64.f);
  U1[(size_t)t * 2 + 1] = im * (1.f / 64.f);
}

__global__ void spec_iy(const float* __restrict__ U1, float* __restrict__ U2) {
  int t = blockIdx.x * blockDim.x + threadIdx.x;
  if (t >= 64 * 64 * 8 * 64) return;
  int d = t & 63, kz = (t >> 6) & 7, y = (t >> 9) & 63, x = t >> 15;
  float re = 0.f, im = 0.f;
  for (int kyi = 0; kyi < 16; ++kyi) {
    int ky = (kyi < 8) ? kyi : kyi + 48;
    const float* s = &U1[(((size_t)(x * 16 + kyi) * 8 + kz) * 64 + d) * 2];
    float ang = TWO_PI * (float)((ky * y) & 63) * (1.f / 64.f);
    float cs = cosf(ang), sn = sinf(ang);
    re += s[0] * cs - s[1] * sn;
    im += s[0] * sn + s[1] * cs;
  }
  U2[(size_t)t * 2 + 0] = re * (1.f / 64.f);
  U2[(size_t)t * 2 + 1] = im * (1.f / 64.f);
}

__global__ void spec_iz(const float* __restrict__ U2, float* __restrict__ SPEC) {
  int t = blockIdx.x * blockDim.x + threadIdx.x;
  if (t >= GRID3 * 64) return;
  int d = t & 63, z = (t >> 6) & 63, p = t >> 12;
  float acc = U2[((size_t)p * 8 * 64 + d) * 2];   // kz=0 real part
  for (int kz = 1; kz < 8; ++kz) {
    const float* s = &U2[(((size_t)p * 8 + kz) * 64 + d) * 2];
    float ang = TWO_PI * (float)((kz * z) & 63) * (1.f / 64.f);
    acc += 2.f * (s[0] * cosf(ang) - s[1] * sinf(ang));
  }
  SPEC[(size_t)t] = acc * (1.f / 64.f);
}

// x = gelu(m2 + x*gw + gb)
__global__ void gate_combine(__bf16* __restrict__ X, const __bf16* __restrict__ M2,
                             const float* __restrict__ gw, const float* __restrict__ gb) {
  int t = blockIdx.x * blockDim.x + threadIdx.x;
  if (t >= GRID3 * 64) return;
  int c = t & 63;
  X[t] = f2bf(gelu_f(bf2f(M2[t]) + bf2f(X[t]) * gw[c] + gb[c]));
}

// edge features: [emb_y(neighbor coords) | emb_x(out coords)] (192 bf16)
__global__ void gno_gather(const float* __restrict__ inp, const float* __restrict__ outp,
                           const int* __restrict__ nb, __bf16* __restrict__ feat,
                           int i0, int npts) {
  int t = blockIdx.x * blockDim.x + threadIdx.x;
  if (t >= npts * KNB * 192) return;
  int comp = t % 192;
  int e = t / 192;
  int j = e % KNB, li = e / KNB;
  int i = i0 + li;
  float coord;
  int local;
  if (comp < 96) {
    local = comp;
    int nbi = nb[(size_t)i * KNB + j];
    coord = inp[(size_t)nbi * 3 + local / 32];
  } else {
    local = comp - 96;
    coord = outp[(size_t)i * 3 + local / 32];
  }
  int inner = local & 31;
  float fr = expf(-9.210340372f * (float)(inner & 15) * (1.f / 16.f));  // ln(1e4)
  float ang = coord * fr;
  feat[(size_t)e * 192 + comp] = f2bf((inner < 16) ? sinf(ang) : cosf(ang));
}

// agg[i,c] = sum_j k[i,j,c] * fy[nb[i,j], c]
__global__ void gno_agg(const __bf16* __restrict__ Kc, const __bf16* __restrict__ X,
                        const int* __restrict__ nb, __bf16* __restrict__ AGG,
                        int i0, int npts) {
  int t = blockIdx.x * blockDim.x + threadIdx.x;
  if (t >= npts * 64) return;
  int c = t & 63, li = t >> 6, i = i0 + li;
  float s = 0.f;
  for (int j = 0; j < KNB; ++j) {
    int nbi = nb[(size_t)i * KNB + j];
    s += bf2f(Kc[((size_t)li * KNB + j) * 64 + c]) * bf2f(X[(size_t)nbi * 64 + c]);
  }
  AGG[(size_t)i * 64 + c] = f2bf(s);
}

// final projection (N=1): out[i] = H[i,:] . W + b
__global__ void proj_out(const __bf16* __restrict__ H, const float* __restrict__ W,
                         const float* __restrict__ b, float* __restrict__ out) {
  int i = blockIdx.x * blockDim.x + threadIdx.x;
  if (i >= NOUT) return;
  float s = b[0];
  for (int c = 0; c < 256; ++c) s += bf2f(H[(size_t)i * 256 + c]) * W[c];
  out[i] = s;
}

// ---------------------------------------------------------------------------
// Host driver
// ---------------------------------------------------------------------------
extern "C" void kernel_launch(void* const* d_in, const int* in_sizes, int n_in,
                              void* d_out, int out_size, void* d_ws, size_t ws_size,
                              hipStream_t stream) {
  (void)in_sizes; (void)n_in; (void)out_size; (void)ws_size;

  // assumed insertion-order recursive flatten of setup_inputs()
  const float* in_p  = (const float*)d_in[0];
  const float* out_p = (const float*)d_in[1];
  const float* f_in  = (const float*)d_in[2];
  const int*   nb    = (const int*)d_in[3];
  const float* liftW[3] = {(const float*)d_in[4], (const float*)d_in[5], (const float*)d_in[6]};
  const float* liftB[3] = {(const float*)d_in[7], (const float*)d_in[8], (const float*)d_in[9]};
  struct FnoP { const float *sw, *skW, *skB, *mW0, *mW1, *mb0, *mb1, *gw, *gb; } fno[4];
  for (int l = 0; l < 4; ++l) {
    int b = 10 + l * 9;
    fno[l].sw  = (const float*)d_in[b + 0];
    fno[l].skW = (const float*)d_in[b + 1];
    fno[l].skB = (const float*)d_in[b + 2];
    fno[l].mW0 = (const float*)d_in[b + 3];
    fno[l].mW1 = (const float*)d_in[b + 4];
    fno[l].mb0 = (const float*)d_in[b + 5];
    fno[l].mb1 = (const float*)d_in[b + 6];
    fno[l].gw  = (const float*)d_in[b + 7];
    fno[l].gb  = (const float*)d_in[b + 8];
  }
  const float* gnoW[3]  = {(const float*)d_in[46], (const float*)d_in[47], (const float*)d_in[48]};
  const float* gnoB[3]  = {(const float*)d_in[49], (const float*)d_in[50], (const float*)d_in[51]};
  const float* projW[2] = {(const float*)d_in[52], (const float*)d_in[53]};
  const float* projB[2] = {(const float*)d_in[54], (const float*)d_in[55]};

  // workspace layout (byte allocator, 256B aligned)
  char* base = (char*)d_ws;
  size_t off = 0;
  auto alloc = [&](size_t bytes) {
    void* p = base + off;
    off = (off + bytes + 255) & ~(size_t)255;
    return p;
  };
  __bf16* Xb   = (__bf16*)alloc((size_t)GRID3 * 64 * 2);
  __bf16* Hb   = (__bf16*)alloc((size_t)GRID3 * 64 * 2);
  __bf16* M2b  = (__bf16*)alloc((size_t)GRID3 * 64 * 2);
  __bf16* HAb  = (__bf16*)alloc((size_t)32768 * 256 * 2);  // also GRID3 x 32
  __bf16* HBb  = (__bf16*)alloc((size_t)32768 * 256 * 2);
  __bf16* IN6b = (__bf16*)alloc((size_t)GRID3 * 32 * 2);
  float*  SPEC = (float*)alloc((size_t)GRID3 * 64 * 4);
  float*  T1   = (float*)alloc((size_t)4096 * 8 * 64 * 2 * 4);
  float*  T2   = (float*)alloc((size_t)64 * 16 * 8 * 64 * 2 * 4);
  float*  T3   = (float*)alloc((size_t)16 * 16 * 8 * 64 * 2 * 4);
  float*  OF   = (float*)alloc((size_t)16 * 16 * 8 * 64 * 2 * 4);
  float*  U1   = (float*)alloc((size_t)64 * 16 * 8 * 64 * 2 * 4);
  float*  U2   = (float*)alloc((size_t)64 * 64 * 8 * 64 * 2 * 4);
  __bf16* FEATb = (__bf16*)alloc((size_t)8000 * 192 * 2);
  __bf16* G1b   = (__bf16*)alloc((size_t)8000 * 512 * 2);
  __bf16* G2b   = (__bf16*)alloc((size_t)8000 * 256 * 2);
  __bf16* KCHb  = (__bf16*)alloc((size_t)8000 * 64 * 2);
  __bf16* AGGb  = (__bf16*)alloc((size_t)NOUT * 64 * 2);
  __bf16* PHb   = (__bf16*)alloc((size_t)NOUT * 256 * 2);
  // transposed bf16 weights: Wt[N][Kpad]
  __bf16* WL0 = (__bf16*)alloc(256 * 32 * 2);
  __bf16* WL1 = (__bf16*)alloc(256 * 256 * 2);
  __bf16* WL2 = (__bf16*)alloc(64 * 256 * 2);
  __bf16* WSK[4], *WM0[4], *WM1[4];
  for (int l = 0; l < 4; ++l) {
    WSK[l] = (__bf16*)alloc(64 * 64 * 2);
    WM0[l] = (__bf16*)alloc(32 * 64 * 2);
    WM1[l] = (__bf16*)alloc(64 * 32 * 2);
  }
  __bf16* WG0 = (__bf16*)alloc(512 * 192 * 2);
  __bf16* WG1 = (__bf16*)alloc(256 * 512 * 2);
  __bf16* WG2 = (__bf16*)alloc(64 * 256 * 2);
  __bf16* WP0 = (__bf16*)alloc(256 * 64 * 2);

  auto conv = [&](const float* s, __bf16* d, int Ks, int Kp, int N) {
    int tot = N * Kp;
    wt_convert<<<(tot + 255) / 256, 256, 0, stream>>>(s, d, Ks, Kp, N);
  };
  conv(liftW[0], WL0, 6, 32, 256);
  conv(liftW[1], WL1, 256, 256, 256);
  conv(liftW[2], WL2, 256, 256, 64);
  for (int l = 0; l < 4; ++l) {
    conv(fno[l].skW, WSK[l], 64, 64, 64);
    conv(fno[l].mW0, WM0[l], 64, 64, 32);
    conv(fno[l].mW1, WM1[l], 32, 32, 64);
  }
  conv(gnoW[0], WG0, 192, 192, 512);
  conv(gnoW[1], WG1, 512, 512, 256);
  conv(gnoW[2], WG2, 256, 256, 64);
  conv(projW[0], WP0, 64, 64, 256);

  auto gemm = [&](const __bf16* A, const __bf16* Wt, const float* bias, const float* res,
                  __bf16* C, int M, int N, int K, int act) {
    if (N % 64 == 0) {
      dim3 g(N / 64, M / 64);
      if (act) gemm_wmma_bf16<1, 4><<<g, 128, 0, stream>>>(A, Wt, bias, res, C, M, N, K);
      else     gemm_wmma_bf16<0, 4><<<g, 128, 0, stream>>>(A, Wt, bias, res, C, M, N, K);
    } else {  // N == 32
      dim3 g(N / 32, M / 64);
      if (act) gemm_wmma_bf16<1, 2><<<g, 128, 0, stream>>>(A, Wt, bias, res, C, M, N, K);
      else     gemm_wmma_bf16<0, 2><<<g, 128, 0, stream>>>(A, Wt, bias, res, C, M, N, K);
    }
  };

  // ---- lift MLP (chunked 8 x 32768 points), K of layer 0 padded to 32 ----
  pack_lift_in<<<(GRID3 * 32 + 255) / 256, 256, 0, stream>>>(f_in, in_p, IN6b, GRID3);
  for (int ch = 0; ch < 8; ++ch) {
    const int Mc = 32768;
    gemm(IN6b + (size_t)ch * Mc * 32, WL0, liftB[0], nullptr, HAb, Mc, 256, 32, 1);
    gemm(HAb, WL1, liftB[1], nullptr, HBb, Mc, 256, 256, 1);
    gemm(HBb, WL2, liftB[2], nullptr, Xb + (size_t)ch * Mc * 64, Mc, 64, 256, 0);
  }

  // ---- 4 FNO blocks ----
  for (int l = 0; l < 4; ++l) {
    spec_fz<<<8192, 256, 0, stream>>>(Xb, T1);
    spec_fy<<<2048, 256, 0, stream>>>(T1, T2);
    spec_fx<<<512, 256, 0, stream>>>(T2, T3);
    spec_modemul<<<512, 256, 0, stream>>>(T3, fno[l].sw, OF);
    spec_ix<<<2048, 256, 0, stream>>>(OF, U1);
    spec_iy<<<8192, 256, 0, stream>>>(U1, U2);
    spec_iz<<<65536, 256, 0, stream>>>(U2, SPEC);
    // h = gelu(spec + x@skip_W + skip_b)
    gemm(Xb, WSK[l], fno[l].skB, SPEC, Hb, GRID3, 64, 64, 1);
    // channel MLP 64->32->64 (HAb reused as [GRID3 x 32])
    gemm(Hb, WM0[l], fno[l].mb0, nullptr, HAb, GRID3, 32, 64, 1);
    gemm(HAb, WM1[l], fno[l].mb1, nullptr, M2b, GRID3, 64, 32, 0);
    // x = gelu(m2 + x*gate_w + gate_b)
    gate_combine<<<65536, 256, 0, stream>>>(Xb, M2b, fno[l].gw, fno[l].gb);
  }

  // ---- GNO (chunked 50 x 800 points = 8000 edges) ----
  for (int q = 0; q < 50; ++q) {
    int i0 = q * 800;
    gno_gather<<<(800 * KNB * 192 + 255) / 256, 256, 0, stream>>>(in_p, out_p, nb, FEATb, i0, 800);
    gemm(FEATb, WG0, gnoB[0], nullptr, G1b, 8000, 512, 192, 1);
    gemm(G1b, WG1, gnoB[1], nullptr, G2b, 8000, 256, 512, 1);
    gemm(G2b, WG2, gnoB[2], nullptr, KCHb, 8000, 64, 256, 0);
    gno_agg<<<(800 * 64 + 255) / 256, 256, 0, stream>>>(KCHb, Xb, nb, AGGb, i0, 800);
  }

  // ---- projection ----
  gemm(AGGb, WP0, projB[0], nullptr, PHb, NOUT, 256, 64, 1);
  proj_out<<<(NOUT + 255) / 256, 256, 0, stream>>>(PHb, projW[1], projB[1], (float*)d_out);
}
